// LSTMModel_86749749444655
// MI455X (gfx1250) — compile-verified, matrix operand order checked by reading, not run on hardware
//
#include <hip/hip_runtime.h>
#include <math.h>

typedef __bf16 bf16_t;
typedef __attribute__((ext_vector_type(16))) __bf16 v16bf;
typedef __attribute__((ext_vector_type(8)))  __bf16 v8bf;
typedef __attribute__((ext_vector_type(8)))  float  v8f;

#define T_  64
#define B_  32
#define P_  9
#define D_  512
#define TB_ (T_ * B_)
#define TD_ (3 * D_)

#if defined(__has_builtin)
#if __has_builtin(__builtin_amdgcn_sched_group_barrier)
#define HAVE_SGB 1
#endif
#endif

// ---------------- small utility kernels ----------------

__global__ void cvt_f32_bf16(const float* __restrict__ in, bf16_t* __restrict__ out, int n) {
  int i = blockIdx.x * 256 + threadIdx.x;
  if (i < n) out[i] = (bf16_t)in[i];
}

__global__ void zero_f32_k(float* p, int n) {
  int i = blockIdx.x * 256 + threadIdx.x;
  if (i < n) p[i] = 0.f;
}

__global__ void zero_i32_k(int* p, int n) {
  int i = blockIdx.x * 256 + threadIdx.x;
  if (i < n) p[i] = 0;
}

__global__ void argmax_k(const float* __restrict__ qmask, int* __restrict__ qi, int* __restrict__ qni) {
  int i = blockIdx.x * 256 + threadIdx.x;
  if (i >= TB_) return;
  int t = i / B_, b = i % B_;
  const float* qm = qmask + ((long)t * B_ + b) * P_;
  int bi = 0; float bv = qm[0];
  for (int p = 1; p < P_; p++) if (qm[p] > bv) { bv = qm[p]; bi = p; }
  qi[i] = bi;
  int tn = (t + 1 < T_) ? t + 1 : T_ - 1;   // qmask_n[t] = qmask[t+1], last repeats
  const float* qn = qmask + ((long)tn * B_ + b) * P_;
  int ci = 0; float cv = qn[0];
  for (int p = 1; p < P_; p++) if (qn[p] > cv) { cv = qn[p]; ci = p; }
  qni[i] = ci;
}

__global__ void count_k(const int* __restrict__ qi, const int* __restrict__ qni,
                        int* cnt_s, int* cnt_l) {
  int i = blockIdx.x * 256 + threadIdx.x;
  if (i >= TB_) return;
  atomicAdd(&cnt_s[qi[i]], 1);
  atomicAdd(&cnt_l[qni[i]], 1);
}

__global__ void prefix_k(const int* cnt_s, const int* cnt_l,
                         int* off_s, int* off_l, int* cur_s, int* cur_l) {
  if (threadIdx.x == 0 && blockIdx.x == 0) {
    int r = 0;
    for (int p = 0; p < P_; p++) { off_s[p] = r; cur_s[p] = r; r += cnt_s[p]; }
    r = 0;
    for (int p = 0; p < P_; p++) { off_l[p] = r; cur_l[p] = r; r += cnt_l[p]; }
  }
}

__global__ void fill_k(const int* __restrict__ qi, const int* __restrict__ qni,
                       int* cur_s, int* cur_l, int* list_s, int* list_l) {
  int i = blockIdx.x * 256 + threadIdx.x;
  if (i >= TB_) return;
  int a = atomicAdd(&cur_s[qi[i]], 1);  list_s[a] = i;
  int b = atomicAdd(&cur_l[qni[i]], 1); list_l[b] = i;
}

// ---------------- WMMA bf16 GEMM: C = rowscale(A @ W^T) + bias ----------------
// A: [M,K] bf16 row-major (optionally gathered rows via per-party sample lists)
// W: [N,K] bf16 row-major (so B-operand = W^T in (K,N) orientation)
// Workgroup: 8 waves = 2(M) x 4(N); wave tile 16x64 (4 x wmma 16x16x32).
// K loop is fully unrolled (KT compile-time) and the pipeline is pinned with
// sched_group_barrier: 2 load-steps in flight, then alternating {4 wmma,
// 10 vmem-read} groups, so each wmma waits only on the previous K-step's
// loads while the current step's loads are outstanding.
template <bool GATHER, int KT>
__global__ __launch_bounds__(256)
void gemm_bf16_k(const bf16_t* __restrict__ A, const bf16_t* __restrict__ Wall,
                 const float* __restrict__ biasAll, const float* __restrict__ scale,
                 float* __restrict__ Cf, bf16_t* __restrict__ Cb,
                 int Min, int N,
                 const int* __restrict__ list, const int* __restrict__ cnts,
                 const int* __restrict__ offs, long wstride) {
  const int tid   = threadIdx.x;
  const int lane  = tid & 31;
  const int wave  = tid >> 5;
  const int nBase = blockIdx.x * 256 + (wave & 3) * 64;
  const int mBase = blockIdx.y * 32 + (wave >> 2) * 16;

  const bf16_t* W   = Wall;
  const float* bias = biasAll;
  int M = Min;
  int grpBase = 0;
  if (GATHER) {
    const int p = blockIdx.z;
    M = cnts[p];
    grpBase = offs[p];
    W = Wall + (long)p * wstride;
    if (biasAll) bias = biasAll + (long)p * N;
  }
  if (mBase >= M || nBase >= N) return;   // wave-uniform

  const int l15   = lane & 15;
  const int khalf = lane >> 4;            // 0: K 0..7/16..23, 1: K 8..15/24..31
  int rowL = mBase + l15;
  if (rowL >= M) rowL = M - 1;            // clamp (partial last tile in gather mode)
  const long aRow = GATHER ? (long)list[grpBase + rowL] * KT : (long)rowL * KT;

  // fragment base pointers (per lane)
  const bf16_t* ap = A + aRow + khalf * 8;
  const bf16_t* wp[4];
#pragma unroll
  for (int ns = 0; ns < 4; ns++)
    wp[ns] = W + (long)(nBase + ns * 16 + l15) * KT + khalf * 8;

  auto loadFrag = [](const bf16_t* p) -> v16bf {
    const v8bf lo = *reinterpret_cast<const v8bf*>(p);
    const v8bf hi = *reinterpret_cast<const v8bf*>(p + 16);
    v16bf r;
#pragma unroll
    for (int i = 0; i < 8; i++) { r[i] = lo[i]; r[8 + i] = hi[i]; }
    return r;
  };

  v8f acc[4] = {};
  v16bf aC = loadFrag(ap);
  v16bf bC[4];
#pragma unroll
  for (int ns = 0; ns < 4; ns++) bC[ns] = loadFrag(wp[ns]);

#pragma unroll
  for (int kk = 0; kk < KT - 32; kk += 32) {
    const int kn = kk + 32;
    // issue next-step loads before consuming current fragments
    v16bf aN = loadFrag(ap + kn);
    v16bf bN[4];
#pragma unroll
    for (int ns = 0; ns < 4; ns++) bN[ns] = loadFrag(wp[ns] + kn);
#pragma unroll
    for (int ns = 0; ns < 4; ns++)
      acc[ns] = __builtin_amdgcn_wmma_f32_16x16x32_bf16(
          false, aC, false, bC[ns], (short)0, acc[ns], false, false);
    aC = aN;
#pragma unroll
    for (int ns = 0; ns < 4; ns++) bC[ns] = bN[ns];
  }
#pragma unroll
  for (int ns = 0; ns < 4; ns++)
    acc[ns] = __builtin_amdgcn_wmma_f32_16x16x32_bf16(
        false, aC, false, bC[ns], (short)0, acc[ns], false, false);

#if HAVE_SGB
  // Pin the software pipeline:  L0 L1 | W0 L2 | W1 L3 | ... | W13 L15 | W14 W15
  // Per K-step: 10 VMEM reads (0x020), 4 WMMA (0x008). KT/32 = 16 steps.
  __builtin_amdgcn_sched_group_barrier(0x020, 20, 0);      // first two load steps
#pragma unroll
  for (int s = 0; s < (KT / 32) - 2; s++) {
    __builtin_amdgcn_sched_group_barrier(0x008, 4, 0);     // wmmas of step s
    __builtin_amdgcn_sched_group_barrier(0x020, 10, 0);    // loads of step s+2
  }
  __builtin_amdgcn_sched_group_barrier(0x008, 8, 0);       // last two wmma steps
#endif

#pragma unroll
  for (int ns = 0; ns < 4; ns++) {
    const int n = nBase + ns * 16 + l15;
    const float bv = bias ? bias[n] : 0.0f;
#pragma unroll
    for (int j = 0; j < 8; j++) {
      const int m = mBase + j + (khalf ? 8 : 0);
      if (m < M) {
        float v = acc[ns][j];
        if (!GATHER && scale) v *= scale[m];   // diag(score)*A commutes -> row scale
        v += bv;
        const long crow = GATHER ? (long)list[grpBase + m] * N : (long)m * N;
        if (Cb) Cb[crow + n] = (bf16_t)v;
        else    Cf[crow + n] = v;
      }
    }
  }
}

// ---------------- per-t batch-axis softmax of <qx,kx> ----------------
__global__ __launch_bounds__(256)
void dot_softmax_k(const bf16_t* __restrict__ qx, const bf16_t* __restrict__ kx,
                   float* __restrict__ score) {
  __shared__ float part[256];
  __shared__ float dotv[B_];
  const int t = blockIdx.x;
  const int tid = threadIdx.x;
  const int b = tid >> 3;
  const int l8 = tid & 7;
  const bf16_t* q = qx + ((long)t * B_ + b) * D_;
  const bf16_t* k = kx + ((long)t * B_ + b) * D_;
  float s = 0.f;
  for (int d = l8; d < D_; d += 8) s += (float)q[d] * (float)k[d];
  part[tid] = s;
  __syncthreads();
  if (tid < B_) {
    float v = 0.f;
#pragma unroll
    for (int i = 0; i < 8; i++) v += part[tid * 8 + i];
    dotv[tid] = v;
  }
  __syncthreads();
  if (tid == 0) {
    float mx = dotv[0];
    for (int i = 1; i < B_; i++) mx = fmaxf(mx, dotv[i]);
    float se = 0.f;
    for (int i = 0; i < B_; i++) { float e = expf(dotv[i] - mx); dotv[i] = e; se += e; }
    const float inv = 1.f / se;
    for (int i = 0; i < B_; i++) score[(long)t * B_ + i] = dotv[i] * inv;
  }
}

// ---------------- serial recurrence step (one WG per batch sample) ----------------
__global__ __launch_bounds__(256)
void step_k(int t, const int* __restrict__ qiArr, const int* __restrict__ qniArr,
            const float* __restrict__ giS, const float* __restrict__ giL,
            const float* __restrict__ Whh, const float* __restrict__ bhh,
            float* __restrict__ S, float* __restrict__ out) {
  __shared__ float hs[D_], hl[D_];
  __shared__ float ghs[TD_], ghl[TD_];
  __shared__ float spk[D_], lst[D_];
  const int b = blockIdx.x;
  const int tid = threadIdx.x;
  const int qi = qiArr[t * B_ + b];
  const int qni = qniArr[t * B_ + b];
  float* sb = S + (long)b * P_ * D_;
  for (int d = tid; d < D_; d += 256) { hs[d] = sb[qi * D_ + d]; hl[d] = sb[qni * D_ + d]; }
  __syncthreads();
  for (int k = tid; k < TD_; k += 256) {
    const float* w1 = Whh + ((long)qi * TD_ + k) * D_;
    const float* w2 = Whh + ((long)qni * TD_ + k) * D_;
    float a1 = 0.f, a2 = 0.f;
    for (int d = 0; d < D_; d++) { a1 += w1[d] * hs[d]; a2 += w2[d] * hl[d]; }
    ghs[k] = a1 + bhh[qi * TD_ + k];
    ghl[k] = a2 + bhh[qni * TD_ + k];
  }
  __syncthreads();
  const long srow = ((long)t * B_ + b) * TD_;
  for (int d = tid; d < D_; d += 256) {
    {
      float ir = giS[srow + d], iz = giS[srow + D_ + d], in = giS[srow + 2 * D_ + d];
      float r = 1.f / (1.f + expf(-(ir + ghs[d])));
      float z = 1.f / (1.f + expf(-(iz + ghs[D_ + d])));
      float n = tanhf(in + r * ghs[2 * D_ + d]);
      spk[d] = (1.f - z) * n + z * hs[d];
    }
    {
      float ir = giL[srow + d], iz = giL[srow + D_ + d], in = giL[srow + 2 * D_ + d];
      float r = 1.f / (1.f + expf(-(ir + ghl[d])));
      float z = 1.f / (1.f + expf(-(iz + ghl[D_ + d])));
      float n = tanhf(in + r * ghl[2 * D_ + d]);
      lst[d] = (1.f - z) * n + z * hl[d];
    }
  }
  __syncthreads();
  float* outRow = out + ((long)(t + 1) * B_ + b) * D_;
  for (int idx = tid; idx < P_ * D_; idx += 256) {
    const int p = idx >> 9;          // D_ == 512
    const int d = idx & (D_ - 1);
    const float cs = (p == qi) ? 1.f : 0.f;
    const float cl = (p == qni) ? 1.f : 0.f;
    const float nv = sb[idx] * (1.f - cs - cl) + cs * spk[d] + cl * lst[d];
    sb[idx] = nv;
    if (p == qi) outRow[d] = nv;     // emit selected speaker state
  }
}

// ---------------- host orchestration ----------------

extern "C" void kernel_launch(void* const* d_in, const int* in_sizes, int n_in,
                              void* d_out, int out_size, void* d_ws, size_t ws_size,
                              hipStream_t stream) {
  (void)in_sizes; (void)n_in; (void)out_size; (void)ws_size;
  const float* U     = (const float*)d_in[0];
  const float* SK    = (const float*)d_in[1];
  const float* NU    = (const float*)d_in[2];
  const float* LK    = (const float*)d_in[3];
  /* d_in[4] = A : unused by the reference */
  const float* qmask = (const float*)d_in[5];
  const float* Wk    = (const float*)d_in[6];
  const float* bk    = (const float*)d_in[7];
  const float* Wq    = (const float*)d_in[8];
  const float* bq    = (const float*)d_in[9];
  const float* Wp    = (const float*)d_in[10];
  const float* bp    = (const float*)d_in[11];
  const float* Wih   = (const float*)d_in[12];
  const float* Whh   = (const float*)d_in[13];
  const float* bih   = (const float*)d_in[14];
  const float* bhh   = (const float*)d_in[15];
  float* out = (float*)d_out;

  char* base = (char*)d_ws;
  size_t off = 0;
  auto carve = [&](size_t bytes) -> void* {
    off = (off + 255) & ~(size_t)255;
    void* p = base + off;
    off += bytes;
    return p;
  };

  bf16_t* Ub   = (bf16_t*)carve((size_t)TB_ * D_ * 2);
  bf16_t* SKb  = (bf16_t*)carve((size_t)TB_ * D_ * 2);
  bf16_t* NUb  = (bf16_t*)carve((size_t)TB_ * D_ * 2);
  bf16_t* LKb  = (bf16_t*)carve((size_t)TB_ * D_ * 2);
  bf16_t* Wkb  = (bf16_t*)carve((size_t)4 * D_ * D_ * 2);
  bf16_t* Wqb  = (bf16_t*)carve((size_t)4 * D_ * D_ * 2);
  bf16_t* Wpb  = (bf16_t*)carve((size_t)4 * D_ * D_ * 2);
  bf16_t* Wihb = (bf16_t*)carve((size_t)P_ * TD_ * D_ * 2);
  bf16_t* Xb   = (bf16_t*)carve((size_t)TB_ * D_ * 2);   // kx scratch
  bf16_t* Yb   = (bf16_t*)carve((size_t)TB_ * D_ * 2);   // qx scratch
  bf16_t* Zb   = (bf16_t*)carve((size_t)TB_ * D_ * 2);   // s_0 / s_1
  bf16_t* Ssb  = (bf16_t*)carve((size_t)TB_ * D_ * 2);   // s_s (all t)
  bf16_t* Slb  = (bf16_t*)carve((size_t)TB_ * D_ * 2);   // s_l (all t)
  float* giS   = (float*)carve((size_t)TB_ * TD_ * 4);
  float* giL   = (float*)carve((size_t)TB_ * TD_ * 4);
  float* score = (float*)carve((size_t)4 * TB_ * 4);
  int* qi      = (int*)carve((size_t)TB_ * 4);
  int* qni     = (int*)carve((size_t)TB_ * 4);
  int* ints    = (int*)carve((size_t)6 * P_ * 4);
  int* cnt_s = ints,        *cnt_l = ints + P_;
  int* off_s = ints + 2*P_, *off_l = ints + 3*P_;
  int* cur_s = ints + 4*P_, *cur_l = ints + 5*P_;
  int* list_s  = (int*)carve((size_t)TB_ * 4);
  int* list_l  = (int*)carve((size_t)TB_ * 4);
  float* S     = (float*)carve((size_t)B_ * P_ * D_ * 4);

  auto cdiv = [](int a, int b) { return (a + b - 1) / b; };
  const int NTB = TB_ * D_;
  const int NW = 4 * D_ * D_;
  const int NWih = P_ * TD_ * D_;

  // bf16 conversions (weights + activations)
  cvt_f32_bf16<<<cdiv(NTB, 256), 256, 0, stream>>>(U, Ub, NTB);
  cvt_f32_bf16<<<cdiv(NTB, 256), 256, 0, stream>>>(SK, SKb, NTB);
  cvt_f32_bf16<<<cdiv(NTB, 256), 256, 0, stream>>>(NU, NUb, NTB);
  cvt_f32_bf16<<<cdiv(NTB, 256), 256, 0, stream>>>(LK, LKb, NTB);
  cvt_f32_bf16<<<cdiv(NW, 256), 256, 0, stream>>>(Wk, Wkb, NW);
  cvt_f32_bf16<<<cdiv(NW, 256), 256, 0, stream>>>(Wq, Wqb, NW);
  cvt_f32_bf16<<<cdiv(NW, 256), 256, 0, stream>>>(Wp, Wpb, NW);
  cvt_f32_bf16<<<cdiv(NWih, 256), 256, 0, stream>>>(Wih, Wihb, NWih);

  zero_f32_k<<<cdiv(B_ * D_, 256), 256, 0, stream>>>(out, B_ * D_);   // outs[0] = 0
  zero_f32_k<<<cdiv(B_ * P_ * D_, 256), 256, 0, stream>>>(S, B_ * P_ * D_);
  zero_i32_k<<<1, 256, 0, stream>>>(ints, 6 * P_);

  // speaker indices + party grouping (for grouped WMMA GRU input projections)
  argmax_k<<<cdiv(TB_, 256), 256, 0, stream>>>(qmask, qi, qni);
  count_k<<<cdiv(TB_, 256), 256, 0, stream>>>(qi, qni, cnt_s, cnt_l);
  prefix_k<<<1, 1, 0, stream>>>(cnt_s, cnt_l, off_s, off_l, cur_s, cur_l);
  fill_k<<<cdiv(TB_, 256), 256, 0, stream>>>(qi, qni, cur_s, cur_l, list_s, list_l);

  auto gemm = [&](const bf16_t* Am, const bf16_t* Wm, const float* biasv,
                  const float* scalev, bf16_t* Cbv) {
    dim3 g(D_ / 256, TB_ / 32, 1);
    gemm_bf16_k<false, D_><<<g, 256, 0, stream>>>(Am, Wm, biasv, scalev, nullptr, Cbv,
                                                  TB_, D_, nullptr, nullptr, nullptr, 0);
  };

  // Attention chain, time-batched (state-independent): speaker side 0 -> 1
  gemm(SKb, Wkb + 0 * D_ * D_, bk + 0 * D_, nullptr, Xb);                  // kx0
  gemm(Ub,  Wqb + 0 * D_ * D_, bq + 0 * D_, nullptr, Yb);                  // qx0
  dot_softmax_k<<<T_, 256, 0, stream>>>(Yb, Xb, score + 0 * TB_);
  gemm(Xb, Wpb + 0 * D_ * D_, bp + 0 * D_, score + 0 * TB_, Zb);           // s_0
  gemm(SKb, Wkb + 1 * D_ * D_, bk + 1 * D_, nullptr, Xb);                  // kx1
  gemm(Zb,  Wqb + 1 * D_ * D_, bq + 1 * D_, nullptr, Yb);                  // qx1
  dot_softmax_k<<<T_, 256, 0, stream>>>(Yb, Xb, score + 1 * TB_);
  gemm(Xb, Wpb + 1 * D_ * D_, bp + 1 * D_, score + 1 * TB_, Ssb);          // s_s
  // listener side 2 -> 3
  gemm(LKb, Wkb + 2 * D_ * D_, bk + 2 * D_, nullptr, Xb);                  // kx2
  gemm(NUb, Wqb + 2 * D_ * D_, bq + 2 * D_, nullptr, Yb);                  // qx2
  dot_softmax_k<<<T_, 256, 0, stream>>>(Yb, Xb, score + 2 * TB_);
  gemm(Xb, Wpb + 2 * D_ * D_, bp + 2 * D_, score + 2 * TB_, Zb);           // s_1
  gemm(LKb, Wkb + 3 * D_ * D_, bk + 3 * D_, nullptr, Xb);                  // kx3
  gemm(Zb,  Wqb + 3 * D_ * D_, bq + 3 * D_, nullptr, Yb);                  // qx3
  dot_softmax_k<<<T_, 256, 0, stream>>>(Yb, Xb, score + 3 * TB_);
  gemm(Xb, Wpb + 3 * D_ * D_, bp + 3 * D_, score + 3 * TB_, Slb);          // s_l

  // GRU input projections: gi = s_* @ Wih[party]^T + bih[party], party-grouped WMMA
  {
    dim3 gg(TD_ / 256, TB_ / 32, P_);
    gemm_bf16_k<true, D_><<<gg, 256, 0, stream>>>(Ssb, Wihb, bih, nullptr, giS, nullptr,
                                                  0, TD_, list_s, cnt_s, off_s,
                                                  (long)TD_ * D_);
    gemm_bf16_k<true, D_><<<gg, 256, 0, stream>>>(Slb, Wihb, bih, nullptr, giL, nullptr,
                                                  0, TD_, list_l, cnt_l, off_l,
                                                  (long)TD_ * D_);
  }

  // Serial recurrence: only Whh @ h is time-dependent. Stream order = dependency.
  for (int t = 0; t < T_; t++)
    step_k<<<B_, 256, 0, stream>>>(t, qi, qni, giS, giL, Whh, bhh, S, out);
}
